// LocalImplicitGrid_53094385713814
// MI455X (gfx1250) — compile-verified, matrix operand order checked by reading, not run on hardware
//
#include <hip/hip_runtime.h>

// ---------------------------------------------------------------------------
// Types / helpers
// ---------------------------------------------------------------------------
typedef __attribute__((ext_vector_type(16))) __bf16 bf16x16;
typedef __attribute__((ext_vector_type(8)))  float  f32x8;

union FragU { uint4 u[2]; bf16x16 v; };

__device__ __forceinline__ unsigned short f2bf(float x) {
    union { float f; unsigned u; } c; c.f = x;
    unsigned r = c.u + 0x7FFFu + ((c.u >> 16) & 1u);   // round-to-nearest-even
    return (unsigned short)(r >> 16);
}
__device__ __forceinline__ float bf2f(unsigned short h) {
    union { unsigned u; float f; } c; c.u = ((unsigned)h) << 16;
    return c.f;
}

#define MTOT   131072          // 16384 points * 8 corners
#define NPTS   16384
#define RCHUNK 8192            // rows per chunk
#define NCHUNK (MTOT / RCHUNK)

// ---------------------------------------------------------------------------
// Weight prep: fp32 W[fi][fo] -> bf16 fragment layout
// dst index = ((nt*KT + kt)*32 + lane)*16 + j
// with n = nt*16 + (lane&15),  k = kt*32 + ((j>>3)<<4) + ((lane>>4)<<3) + (j&7)
// ---------------------------------------------------------------------------
__global__ void wprep_kernel(const float* __restrict__ W,
                             unsigned short* __restrict__ dst,
                             int fi, int fo, int KT, int total) {
    int idx = blockIdx.x * blockDim.x + threadIdx.x;
    if (idx >= total) return;
    int j    = idx & 15;
    int lane = (idx >> 4) & 31;
    int t    = idx >> 9;
    int kt   = t % KT;
    int nt   = t / KT;
    int k = kt * 32 + ((j >> 3) << 4) + ((lane >> 4) << 3) + (j & 7);
    int n = nt * 16 + (lane & 15);
    dst[idx] = (k < fi) ? f2bf(W[(size_t)k * fo + n]) : (unsigned short)0;
}

// ---------------------------------------------------------------------------
// Gather: per (point, corner) row -> X[row][32] bf16 (xloc3 + lat16 + 0 pad),
// and trilinear weight wc[row].
// ---------------------------------------------------------------------------
__global__ void gather_kernel(const float* __restrict__ grid,
                              const float* __restrict__ pts,
                              unsigned short* __restrict__ X,
                              float* __restrict__ wc) {
    int idx = blockIdx.x * blockDim.x + threadIdx.x;
    if (idx >= MTOT) return;
    int pi = idx >> 3, j = idx & 7;

    float px = fminf(fmaxf(pts[pi * 3 + 0], 1e-6f), 1.f - 1e-6f) * 31.f;
    float py = fminf(fmaxf(pts[pi * 3 + 1], 1e-6f), 1.f - 1e-6f) * 31.f;
    float pz = fminf(fmaxf(pts[pi * 3 + 2], 1e-6f), 1.f - 1e-6f) * 31.f;
    float i0x = floorf(px), i0y = floorf(py), i0z = floorf(pz);

    int cx = (j >> 2) & 1, cy = (j >> 1) & 1, cz = j & 1;

    float xl0 = px - (i0x + (float)cx);
    float xl1 = py - (i0y + (float)cy);
    float xl2 = pz - (i0z + (float)cz);

    float wx = fabsf(px - (i0x + 1.f - (float)cx));
    float wy = fabsf(py - (i0y + 1.f - (float)cy));
    float wz = fabsf(pz - (i0z + 1.f - (float)cz));
    wc[idx] = wx * wy * wz;

    int ix = (int)i0x + cx, iy = (int)i0y + cy, iz = (int)i0z + cz;
    const float* lat = grid + ((((size_t)ix * 32 + iy) * 32) + iz) * 16;

    unsigned short* xr = X + (size_t)idx * 32;
    xr[0] = f2bf(xl0); xr[1] = f2bf(xl1); xr[2] = f2bf(xl2);
    #pragma unroll
    for (int c = 0; c < 16; ++c) xr[3 + c] = f2bf(lat[c]);
    #pragma unroll
    for (int c = 19; c < 32; ++c) xr[c] = 0;
}

// ---------------------------------------------------------------------------
// Tail writer: copy 32-col padded x (X row) into the skip-concat tail of each
// intermediate activation buffer for one chunk of rows.
// ---------------------------------------------------------------------------
__global__ void tail_kernel(const unsigned short* __restrict__ X, int rowBase,
                            unsigned short* __restrict__ b1,
                            unsigned short* __restrict__ b2,
                            unsigned short* __restrict__ b3,
                            unsigned short* __restrict__ b4) {
    int r = blockIdx.x * blockDim.x + threadIdx.x;
    if (r >= RCHUNK) return;
    const uint4* src = (const uint4*)(X + (size_t)(rowBase + r) * 32);
    uint4 v0 = src[0], v1 = src[1], v2 = src[2], v3 = src[3];
    uint4* d1 = (uint4*)(b1 + (size_t)r * 2080 + 2048);
    uint4* d2 = (uint4*)(b2 + (size_t)r * 1056 + 1024);
    uint4* d3 = (uint4*)(b3 + (size_t)r *  544 +  512);
    uint4* d4 = (uint4*)(b4 + (size_t)r *  288 +  256);
    d1[0] = v0; d1[1] = v1; d1[2] = v2; d1[3] = v3;
    d2[0] = v0; d2[1] = v1; d2[2] = v2; d2[3] = v3;
    d3[0] = v0; d3[1] = v1; d3[2] = v2; d3[3] = v3;
    d4[0] = v0; d4[1] = v1; d4[2] = v2; d4[3] = v3;
}

// ---------------------------------------------------------------------------
// WMMA GEMM: C[256m x 128n per block] = act(A @ W + bias), bf16 in, f32 acc.
// 256 threads = 8 waves; wave w owns rows {w*16..} and {128+w*16..}, all 8
// n-tiles -> 16 wmma per K-step per wave, each B fragment reused 2x.
// A tiles double-buffered in LDS (one barrier per K-step); B fragments
// streamed from pre-swizzled L2-resident weights through a 4-deep register
// ring (load-to-use distance = 4 tiles). KT is a template constant so all B
// fragment addresses fold into 24-bit instruction immediate offsets.
// ---------------------------------------------------------------------------
template <int KT>
__global__ __launch_bounds__(256)
void gemm_bf16(const unsigned short* __restrict__ A, int ldA,
               const unsigned short* __restrict__ Wf,
               const float* __restrict__ bias,
               unsigned short* __restrict__ C, int ldC, int applyAct) {
    __shared__ unsigned short As[2][256 * 32];

    const int tid  = threadIdx.x;
    const int w    = tid >> 5;
    const int lane = tid & 31;
    const int mBlock  = blockIdx.y * 256;
    const int nBlockT = blockIdx.x * 8;        // n offset in 16-col tiles

    f32x8 acc[2][8];
    #pragma unroll
    for (int h = 0; h < 2; ++h)
        #pragma unroll
        for (int i = 0; i < 8; ++i) acc[h][i] = (f32x8)(0.f);

    // A staging: thread t loads all 32 cols of row (mBlock + t) per K-step.
    const unsigned short* aStage = A + (size_t)(mBlock + tid) * ldA;

    // A fragment read positions (ISA 16-bit A layout)
    const int aRow = w * 16 + (lane & 15);
    const int kb   = (lane >> 4) << 3;
    const int fOff0 = aRow * 32 + kb;            // half 0
    const int fOff1 = (128 + aRow) * 32 + kb;    // half 1
    const int sOff  = tid * 32;

    constexpr size_t BSTR = (size_t)KT * 512;    // elements between n-tiles
    const unsigned short* bCol = Wf + ((size_t)nBlockT * KT * 32 + lane) * 16;

    // preload K-tile 0 into LDS buffer 0
    {
        const uint4* g = (const uint4*)aStage;
        uint4 t0 = g[0], t1 = g[1], t2 = g[2], t3 = g[3];
        uint4* s = (uint4*)&As[0][sOff];
        s[0] = t0; s[1] = t1; s[2] = t2; s[3] = t3;
    }
    __syncthreads();

    int buf = 0;
    for (int kt = 0; kt < KT; ++kt) {
        // issue next A tile's global loads early
        uint4 n0, n1, n2, n3;
        const bool hasNext = (kt + 1 < KT);
        if (hasNext) {
            const uint4* g = (const uint4*)(aStage + (size_t)(kt + 1) * 32);
            n0 = g[0]; n1 = g[1]; n2 = g[2]; n3 = g[3];
        }

        // A fragments for both 128-row halves
        FragU af0, af1;
        af0.u[0] = *(const uint4*)(&As[buf][fOff0]);
        af0.u[1] = *(const uint4*)(&As[buf][fOff0 + 16]);
        af1.u[0] = *(const uint4*)(&As[buf][fOff1]);
        af1.u[1] = *(const uint4*)(&As[buf][fOff1 + 16]);

        // B fragments: 4-deep register ring; constant-offset loads.
        const unsigned short* bKt = bCol + (size_t)kt * 512;
        FragU bring[4];
        #pragma unroll
        for (int i = 0; i < 4; ++i)
            bring[i].v = *(const bf16x16*)(bKt + (size_t)i * BSTR);

        #pragma unroll
        for (int nt = 0; nt < 8; ++nt) {
            acc[0][nt] = __builtin_amdgcn_wmma_f32_16x16x32_bf16(
                false, af0.v, false, bring[nt & 3].v, (short)0,
                acc[0][nt], false, false);
            acc[1][nt] = __builtin_amdgcn_wmma_f32_16x16x32_bf16(
                false, af1.v, false, bring[nt & 3].v, (short)0,
                acc[1][nt], false, false);
            if (nt + 4 < 8)
                bring[nt & 3].v =
                    *(const bf16x16*)(bKt + (size_t)(nt + 4) * BSTR);
        }

        // store next tile to the other buffer (disjoint from current readers)
        if (hasNext) {
            uint4* s = (uint4*)&As[buf ^ 1][sOff];
            s[0] = n0; s[1] = n1; s[2] = n2; s[3] = n3;
            __syncthreads();                    // publish before next K-step
        }
        buf ^= 1;
    }

    // Epilogue: bias + leaky relu + bf16 store (ISA 32-bit C/D layout)
    const int nCol0 = (nBlockT << 4) + (lane & 15);
    #pragma unroll
    for (int h = 0; h < 2; ++h) {
        const int mOff = mBlock + h * 128 + w * 16 + ((lane >> 4) << 3);
        #pragma unroll
        for (int nt = 0; nt < 8; ++nt) {
            int n = nCol0 + nt * 16;
            float b = bias[n];
            #pragma unroll
            for (int r = 0; r < 8; ++r) {
                float v = acc[h][nt][r] + b;
                if (applyAct) v = fmaxf(v, 0.f) + 0.2f * fminf(v, 0.f);
                C[(size_t)(mOff + r) * ldC + n] = f2bf(v);
            }
        }
    }
}

// ---------------------------------------------------------------------------
// Final: fc5 (128->1) + corner-weighted sum over 8 corners per point.
// ---------------------------------------------------------------------------
__global__ void final_kernel(const unsigned short* __restrict__ h5,
                             const float* __restrict__ wc, int rowBase,
                             const float* __restrict__ W5,
                             const float* __restrict__ b5,
                             float* __restrict__ out) {
    int p = blockIdx.x * blockDim.x + threadIdx.x;   // local point in chunk
    if (p >= RCHUNK / 8) return;
    float sum = 0.f;
    #pragma unroll 1
    for (int j = 0; j < 8; ++j) {
        int rLocal = p * 8 + j;
        const unsigned short* h = h5 + (size_t)rLocal * 128;
        float d = 0.f;
        for (int c = 0; c < 128; ++c) d += bf2f(h[c]) * W5[c];
        d += b5[0];
        sum += wc[rowBase + rLocal] * d;
    }
    out[rowBase / 8 + p] = sum;
}

// ---------------------------------------------------------------------------
// Host side
// ---------------------------------------------------------------------------
extern "C" void kernel_launch(void* const* d_in, const int* in_sizes, int n_in,
                              void* d_out, int out_size, void* d_ws, size_t ws_size,
                              hipStream_t stream) {
    const float* grid = (const float*)d_in[0];
    const float* pts  = (const float*)d_in[1];
    const float* W[6]; const float* bv[6];
    for (int i = 0; i < 6; ++i) { W[i] = (const float*)d_in[2 + 2 * i]; bv[i] = (const float*)d_in[3 + 2 * i]; }
    float* out = (float*)d_out;

    // layer dims (fi, fo) and K-tiles (Kpad/32)
    const int fi[5] = { 19, 2067, 1043, 531, 275 };
    const int fo[5] = { 2048, 1024, 512, 256, 128 };
    const int KT[5] = { 1, 65, 33, 17, 9 };

    // workspace layout (all sizes multiples of 256B)
    char* ws = (char*)d_ws;
    unsigned short* Wf[5];
    size_t off = 0;
    size_t wfEls[5];
    for (int i = 0; i < 5; ++i) {
        wfEls[i] = (size_t)(fo[i] / 16) * KT[i] * 512;
        Wf[i] = (unsigned short*)(ws + off);
        off += wfEls[i] * 2;
    }
    unsigned short* X = (unsigned short*)(ws + off); off += (size_t)MTOT * 32 * 2;
    float* wc         = (float*)(ws + off);          off += (size_t)MTOT * 4;
    unsigned short* buf1 = (unsigned short*)(ws + off); off += (size_t)RCHUNK * 2080 * 2;
    unsigned short* buf2 = (unsigned short*)(ws + off); off += (size_t)RCHUNK * 1056 * 2;
    unsigned short* buf3 = (unsigned short*)(ws + off); off += (size_t)RCHUNK * 544 * 2;
    unsigned short* buf4 = (unsigned short*)(ws + off); off += (size_t)RCHUNK * 288 * 2;
    unsigned short* buf5 = (unsigned short*)(ws + off); off += (size_t)RCHUNK * 128 * 2;

    // 1) weight fragment prep
    for (int i = 0; i < 5; ++i) {
        int total = (int)wfEls[i];
        wprep_kernel<<<(total + 255) / 256, 256, 0, stream>>>(W[i], Wf[i], fi[i], fo[i], KT[i], total);
    }
    // 2) gather inputs
    gather_kernel<<<MTOT / 256, 256, 0, stream>>>(grid, pts, X, wc);

    // 3) chunked MLP
    for (int c = 0; c < NCHUNK; ++c) {
        int rowBase = c * RCHUNK;
        tail_kernel<<<RCHUNK / 256, 256, 0, stream>>>(X, rowBase, buf1, buf2, buf3, buf4);

        // fc0: X(32) -> buf1[0:2048)
        gemm_bf16<1><<<dim3(2048 / 128, RCHUNK / 256), 256, 0, stream>>>(
            X + (size_t)rowBase * 32, 32, Wf[0], bv[0], buf1, 2080, 1);
        // fc1: buf1(2080) -> buf2[0:1024)
        gemm_bf16<65><<<dim3(1024 / 128, RCHUNK / 256), 256, 0, stream>>>(
            buf1, 2080, Wf[1], bv[1], buf2, 1056, 1);
        // fc2: buf2(1056) -> buf3[0:512)
        gemm_bf16<33><<<dim3(512 / 128, RCHUNK / 256), 256, 0, stream>>>(
            buf2, 1056, Wf[2], bv[2], buf3, 544, 1);
        // fc3: buf3(544) -> buf4[0:256)
        gemm_bf16<17><<<dim3(256 / 128, RCHUNK / 256), 256, 0, stream>>>(
            buf3, 544, Wf[3], bv[3], buf4, 288, 1);
        // fc4: buf4(288) -> buf5[0:128)
        gemm_bf16<9><<<dim3(128 / 128, RCHUNK / 256), 256, 0, stream>>>(
            buf4, 288, Wf[4], bv[4], buf5, 128, 1);
        // fc5 + trilinear sum
        final_kernel<<<(RCHUNK / 8 + 255) / 256, 256, 0, stream>>>(
            buf5, wc, rowBase, W[5], bv[5], out);
    }
}
